// ProteinGN_24438363914612
// MI455X (gfx1250) — compile-verified, matrix operand order checked by reading, not run on hardware
//
#include <hip/hip_runtime.h>
#include <math.h>

typedef __attribute__((ext_vector_type(16))) _Float16 v16h;
typedef __attribute__((ext_vector_type(8)))  float    v8f;

#define N_NODES 100000
#define N_EDGES 1000000
#define NB      32
#define EPG     31250     // undirected edges per graph
#define NPG     3125      // nodes per graph

// ---------------------------------------------------------------------------
// Prep: fold g=relu(g_b) into biases; pack f16 weight fragments in WMMA B layout.
// Fragment element (lane l, elem i) holds W[k][n] with
//   k = (i&7) + 8*(l>>4) + 16*(i>>3)  (+32*kt),  n = 16*nt + (l&15)
// mirroring the documented 16-bit A layout.
// ---------------------------------------------------------------------------
__global__ void prep_kernel(const float* __restrict__ g_b,
                            const float* __restrict__ le_We, const float* __restrict__ le_Ws,
                            const float* __restrict__ le_Wg, const float* __restrict__ le_b,
                            const float* __restrict__ ln_Wn, const float* __restrict__ ln_Wi,
                            const float* __restrict__ ln_Wg, const float* __restrict__ ln_b,
                            _Float16* __restrict__ Wle, _Float16* __restrict__ Wln,
                            float* __restrict__ bias_e, float* __restrict__ bias_n)
{
    int tid = threadIdx.x;
    float g0 = fmaxf(g_b[0], 0.f), g1 = fmaxf(g_b[1], 0.f);
    float g2 = fmaxf(g_b[2], 0.f), g3 = fmaxf(g_b[3], 0.f);
    for (int c = tid; c < 64; c += blockDim.x)
        bias_e[c] = le_b[c] + g0*le_Wg[c] + g1*le_Wg[64+c] + g2*le_Wg[128+c] + g3*le_Wg[192+c];
    for (int c = tid; c < 128; c += blockDim.x)
        bias_n[c] = ln_b[c] + g0*ln_Wg[c] + g1*ln_Wg[128+c] + g2*ln_Wg[256+c] + g3*ln_Wg[384+c];
    // Wle: K=32 (rows 0-7 le_We, 8-23 le_Ws, rest 0), NT=4
    for (int idx = tid; idx < 4*32*16; idx += blockDim.x) {
        int i = idx & 15, l = (idx >> 4) & 31, nt = idx >> 9;
        int k = (i & 7) + 8*(l >> 4) + 16*(i >> 3);
        int n = nt*16 + (l & 15);
        float w = 0.f;
        if (k < 8)       w = le_We[k*64 + n];
        else if (k < 24) w = le_Ws[(k-8)*64 + n];
        Wle[idx] = (_Float16)w;
    }
    // Wln: K=96 (rows 0-15 ln_Wn, 16-79 ln_Wi, rest 0), KT=3, NT=8
    for (int idx = tid; idx < 24*32*16; idx += blockDim.x) {
        int i = idx & 15, l = (idx >> 4) & 31, f = idx >> 9;
        int nt = f & 7, kt = f >> 3;
        int k = 32*kt + (i & 7) + 8*(l >> 4) + 16*(i >> 3);
        int n = nt*16 + (l & 15);
        float w = 0.f;
        if (k < 16)      w = ln_Wn[k*128 + n];
        else if (k < 80) w = ln_Wi[(k-16)*128 + n];
        Wln[idx] = (_Float16)w;
    }
}

// ---------------------------------------------------------------------------
// Edge featurize + encoder (23->4->8), once per undirected edge, output f16.
// ---------------------------------------------------------------------------
__global__ void edge_enc_kernel(const float* __restrict__ distances,
                                const int* __restrict__ senders, const int* __restrict__ receivers,
                                const float* __restrict__ e_W1, const float* __restrict__ e_b1,
                                const float* __restrict__ e_W2, const float* __restrict__ e_b2,
                                _Float16* __restrict__ e8)
{
    int e = blockIdx.x*blockDim.x + threadIdx.x;
    if (e >= N_EDGES) return;
    float d = distances[e];
    float rbf[16];
#pragma unroll
    for (int j = 0; j < 16; ++j) { float t = d - (20.f/15.f)*j; rbf[j] = __expf(-t*t); }
    float sep = (float)(senders[e] - receivers[e] + 1);
    const float bins[7] = {-10.f,-5.f,-4.f,-3.f,-2.f,-1.f,0.f};
    int cnt = 0;
#pragma unroll
    for (int j = 0; j < 7; ++j) cnt += (bins[j] < sep) ? 1 : 0;
    int cls = 6 - cnt; cls = cls < 0 ? 0 : (cls > 6 ? 6 : cls);
    float h1[4];
#pragma unroll
    for (int c = 0; c < 4; ++c) {
        float acc = e_b1[c] + e_W1[(16+cls)*4 + c];
#pragma unroll
        for (int j = 0; j < 16; ++j) acc += rbf[j]*e_W1[j*4 + c];
        h1[c] = fmaxf(acc, 0.f);
    }
#pragma unroll
    for (int c = 0; c < 8; ++c) {
        float acc = e_b2[c];
#pragma unroll
        for (int j = 0; j < 4; ++j) acc += h1[j]*e_W2[j*8 + c];
        e8[(size_t)e*8 + c] = (_Float16)fmaxf(acc, 0.f);
    }
}

// ---------------------------------------------------------------------------
// Node encoder (83->8->16), output f16.
// ---------------------------------------------------------------------------
__global__ void node_enc_kernel(const float* __restrict__ node_features,
                                const int* __restrict__ residues, const float* __restrict__ emb,
                                const float* __restrict__ n_W1, const float* __restrict__ n_b1,
                                const float* __restrict__ n_W2, const float* __restrict__ n_b2,
                                _Float16* __restrict__ x16)
{
    int n = blockIdx.x*blockDim.x + threadIdx.x;
    if (n >= N_NODES) return;
    const float* em = emb + residues[n]*32;
    const float* nf = node_features + (size_t)n*51;
    float h1[8];
#pragma unroll
    for (int c = 0; c < 8; ++c) h1[c] = n_b1[c];
    for (int j = 0; j < 32; ++j) {
        float v = em[j];
#pragma unroll
        for (int c = 0; c < 8; ++c) h1[c] += v*n_W1[j*8 + c];
    }
    for (int j = 0; j < 51; ++j) {
        float v = nf[j];
#pragma unroll
        for (int c = 0; c < 8; ++c) h1[c] += v*n_W1[(32+j)*8 + c];
    }
#pragma unroll
    for (int c = 0; c < 8; ++c) h1[c] = fmaxf(h1[c], 0.f);
#pragma unroll
    for (int c = 0; c < 16; ++c) {
        float acc = n_b2[c];
#pragma unroll
        for (int j = 0; j < 8; ++j) acc += h1[j]*n_W2[j*16 + c];
        x16[(size_t)n*16 + c] = (_Float16)fmaxf(acc, 0.f);
    }
}

// ---------------------------------------------------------------------------
// Edge MLP via WMMA: one wave = 16 directed edges, A=[e8|x16[s]] (K=24 pad 32),
// D = 16x64 via 4x v_wmma_f32_16x16x32_f16. Bias preloaded in C. Scatter to
// inc_sum[r] and per-graph e_sum (wave-reduced).
// ---------------------------------------------------------------------------
__global__ void __launch_bounds__(256) edge_mlp_kernel(
    const _Float16* __restrict__ e8, const _Float16* __restrict__ x16,
    const int* __restrict__ senders, const int* __restrict__ receivers,
    const _Float16* __restrict__ Wle, const float* __restrict__ bias_e,
    float* __restrict__ inc_sum, float* __restrict__ e_sum, float* __restrict__ deg)
{
    int lane = threadIdx.x & 31;
    int tile = (blockIdx.x*blockDim.x + threadIdx.x) >> 5;   // 125000 tiles exactly
    int hl = lane >> 4, col = lane & 15;

    int de = tile*16 + col;
    int ei = de < N_EDGES ? de : de - N_EDGES;
    int s, r;
    if (de < N_EDGES) { s = senders[ei];   r = receivers[ei]; }
    else              { s = receivers[ei]; r = senders[ei];   }

    // A fragment: lanes 0-15 hold K 0..7 (e8) and 16..23 (x[8..15]);
    //             lanes 16-31 hold K 8..15 (x[0..7]) and 24..31 (pad 0).
    v16h a;
    const _Float16* xr = x16 + (size_t)s*16;
    if (hl == 0) {
        const _Float16* er = e8 + (size_t)ei*8;
#pragma unroll
        for (int i = 0; i < 8; ++i) a[i] = er[i];
#pragma unroll
        for (int i = 0; i < 8; ++i) a[8+i] = xr[8+i];
    } else {
#pragma unroll
        for (int i = 0; i < 8; ++i) a[i] = xr[i];
#pragma unroll
        for (int i = 0; i < 8; ++i) a[8+i] = (_Float16)0.0f;
    }

    v8f acc[4];
#pragma unroll
    for (int nt = 0; nt < 4; ++nt) {
        float bn = bias_e[nt*16 + col];
        v8f c0 = { bn,bn,bn,bn,bn,bn,bn,bn };
        v16h b = *(const v16h*)(Wle + ((size_t)(nt*32 + lane))*16);
        acc[nt] = __builtin_amdgcn_wmma_f32_16x16x32_f16(false, a, false, b, (short)0, c0, false, false);
    }
#pragma unroll
    for (int nt = 0; nt < 4; ++nt)
#pragma unroll
        for (int v = 0; v < 8; ++v) acc[nt][v] = fmaxf(acc[nt][v], 0.f);

    if (hl == 0) atomicAdd(&deg[r], 1.0f);   // lanes 0..15 own edge m=col

    // scatter: D layout -> lane's acc[nt][v] belongs to edge m=v+8*hl, feature nt*16+col
    int de0 = tile*16;
    int ei0 = de0 < N_EDGES ? de0 : de0 - N_EDGES;
    int gfirst = ei0 / EPG;
    float gacc[4] = {0.f, 0.f, 0.f, 0.f};
#pragma unroll
    for (int v = 0; v < 8; ++v) {
        int de2 = tile*16 + v + 8*hl;
        int ei2 = de2 < N_EDGES ? de2 : de2 - N_EDGES;
        int rr  = (de2 < N_EDGES) ? receivers[ei2] : senders[ei2];
        int gg  = ei2 / EPG;
#pragma unroll
        for (int nt = 0; nt < 4; ++nt) {
            float hv = acc[nt][v];
            atomicAdd(&inc_sum[(size_t)rr*64 + nt*16 + col], hv);
            if (gg == gfirst) gacc[nt] += hv;
            else              atomicAdd(&e_sum[gg*64 + nt*16 + col], hv);
        }
    }
#pragma unroll
    for (int nt = 0; nt < 4; ++nt)
        atomicAdd(&e_sum[gfirst*64 + nt*16 + col], gacc[nt]);
}

// ---------------------------------------------------------------------------
// Node MLP via WMMA (A=[x16|inc], K=80 pad 96, 3x8 WMMAs per 16-node tile),
// fused sigmoid readout from LDS + block-reduced n_sum.
// ---------------------------------------------------------------------------
__global__ void __launch_bounds__(256) node_mlp_kernel(
    const _Float16* __restrict__ x16, const float* __restrict__ inc_sum,
    const float* __restrict__ deg, const _Float16* __restrict__ Wln,
    const float* __restrict__ bias_n, const float* __restrict__ ro_nW,
    const float* __restrict__ ro_nb, float* __restrict__ n_sum,
    float* __restrict__ out_nodes)
{
    __shared__ float xh[128*128];   // 64KB: 128 nodes x 128 features
    int lane = threadIdx.x & 31;
    int wib  = threadIdx.x >> 5;                 // wave in block (0..7)
    int tile = blockIdx.x*8 + wib;               // 6250 tiles total
    int hl = lane >> 4, col = lane & 15;

    if (tile < N_NODES/16) {
        int node = tile*16 + col;
        const _Float16* xr  = x16 + (size_t)node*16;
        const float* incp   = inc_sum + (size_t)node*64;
        float rd = 1.f/fmaxf(deg[node], 1.f);

        v8f acc[8];
#pragma unroll
        for (int nt = 0; nt < 8; ++nt) {
            float bn = bias_n[nt*16 + col];
            v8f c0 = { bn,bn,bn,bn,bn,bn,bn,bn };
            acc[nt] = c0;
        }
#pragma unroll
        for (int kt = 0; kt < 3; ++kt) {
            v16h a;
            int kA = 32*kt + 8*hl;
            int kB = 32*kt + 16 + 8*hl;
#pragma unroll
            for (int i = 0; i < 8; ++i) {
                int k = kA + i;
                a[i] = (k < 16) ? xr[k] : ((k < 80) ? (_Float16)(incp[k-16]*rd) : (_Float16)0.0f);
            }
#pragma unroll
            for (int i = 0; i < 8; ++i) {
                int k = kB + i;
                a[8+i] = (k < 16) ? xr[k] : ((k < 80) ? (_Float16)(incp[k-16]*rd) : (_Float16)0.0f);
            }
#pragma unroll
            for (int nt = 0; nt < 8; ++nt) {
                v16h b = *(const v16h*)(Wln + ((size_t)((kt*8 + nt)*32 + lane))*16);
                acc[nt] = __builtin_amdgcn_wmma_f32_16x16x32_f16(false, a, false, b, (short)0, acc[nt], false, false);
            }
        }
#pragma unroll
        for (int nt = 0; nt < 8; ++nt)
#pragma unroll
            for (int v = 0; v < 8; ++v)
                xh[(wib*16 + v + 8*hl)*128 + nt*16 + col] = fmaxf(acc[nt][v], 0.f);
    }
    __syncthreads();

    int tid = threadIdx.x;
    int nodebase = blockIdx.x*128;
    if (tid < 128) {                              // per-node sigmoid readout
        int node = nodebase + tid;
        if (node < N_NODES) {
            float d0 = ro_nb[0], d1 = ro_nb[1];
            const float* row = &xh[tid*128];
            for (int f = 0; f < 128; ++f) {
                float v = row[f];
                d0 += v*ro_nW[f*2+0];
                d1 += v*ro_nW[f*2+1];
            }
            out_nodes[(size_t)node*2+0] = 1.f/(1.f + __expf(-d0));
            out_nodes[(size_t)node*2+1] = 1.f/(1.f + __expf(-d1));
        }
    }
    {                                             // block-level n_sum reduction
        int f  = tid & 127;
        int r0 = (tid < 128) ? 0 : 64;
        if (nodebase + r0 < N_NODES) {
            int gfirst = (nodebase + r0) / NPG;
            float accv = 0.f;
            for (int rr = r0; rr < r0 + 64; ++rr) {
                int gn = nodebase + rr;
                if (gn >= N_NODES) break;
                int g = gn / NPG;
                float v = xh[rr*128 + f];
                if (g == gfirst) accv += v;
                else             atomicAdd(&n_sum[g*128 + f], v);
            }
            atomicAdd(&n_sum[gfirst*128 + f], accv);
        }
    }
}

// ---------------------------------------------------------------------------
// Global head: gh = relu(e_mean@lg_We + n_mean@lg_Wn + g@lg_Wg + lg_b),
// global_out = sigmoid(gh@ro_gW + ro_gb). ne=62500, nn=3125 are constants.
// ---------------------------------------------------------------------------
__global__ void finalize_kernel(const float* __restrict__ e_sum, const float* __restrict__ n_sum,
                                const float* __restrict__ g_b,
                                const float* __restrict__ lg_We, const float* __restrict__ lg_Wn,
                                const float* __restrict__ lg_Wg, const float* __restrict__ lg_b,
                                const float* __restrict__ ro_gW, const float* __restrict__ ro_gb,
                                float* __restrict__ out_glob)
{
    __shared__ float gh[32];
    int b = blockIdx.x;
    int n = threadIdx.x;                          // 0..31
    float acc = lg_b[n];
    for (int k = 0; k < 64; ++k)  acc += (e_sum[b*64 + k]  * (1.f/62500.f)) * lg_We[k*32 + n];
    for (int k = 0; k < 128; ++k) acc += (n_sum[b*128 + k] * (1.f/3125.f))  * lg_Wn[k*32 + n];
#pragma unroll
    for (int j = 0; j < 4; ++j)   acc += fmaxf(g_b[j], 0.f) * lg_Wg[j*32 + n];
    gh[n] = fmaxf(acc, 0.f);
    __syncthreads();
    if (n < 5) {
        float o = ro_gb[n];
        for (int k = 0; k < 32; ++k) o += gh[k]*ro_gW[k*5 + n];
        out_glob[b*5 + n] = 1.f/(1.f + __expf(-o));
    }
}

extern "C" void kernel_launch(void* const* d_in, const int* in_sizes, int n_in,
                              void* d_out, int out_size, void* d_ws, size_t ws_size,
                              hipStream_t stream)
{
    (void)in_sizes; (void)n_in; (void)out_size; (void)ws_size;
    const float* node_features = (const float*)d_in[0];
    const float* distances    = (const float*)d_in[1];
    const int*   residues     = (const int*)  d_in[2];
    const int*   senders      = (const int*)  d_in[3];
    const int*   receivers    = (const int*)  d_in[4];
    const float* emb  = (const float*)d_in[7];
    const float* e_W1 = (const float*)d_in[8];
    const float* e_b1 = (const float*)d_in[9];
    const float* e_W2 = (const float*)d_in[10];
    const float* e_b2 = (const float*)d_in[11];
    const float* n_W1 = (const float*)d_in[12];
    const float* n_b1 = (const float*)d_in[13];
    const float* n_W2 = (const float*)d_in[14];
    const float* n_b2 = (const float*)d_in[15];
    const float* g_b  = (const float*)d_in[16];
    const float* le_We = (const float*)d_in[17];
    const float* le_Ws = (const float*)d_in[18];
    const float* le_Wg = (const float*)d_in[19];
    const float* le_b  = (const float*)d_in[20];
    const float* ln_Wn = (const float*)d_in[21];
    const float* ln_Wi = (const float*)d_in[22];
    const float* ln_Wg = (const float*)d_in[23];
    const float* ln_b  = (const float*)d_in[24];
    const float* lg_We = (const float*)d_in[25];
    const float* lg_Wn = (const float*)d_in[26];
    const float* lg_Wg = (const float*)d_in[27];
    const float* lg_b  = (const float*)d_in[28];
    const float* ro_nW = (const float*)d_in[29];
    const float* ro_nb = (const float*)d_in[30];
    const float* ro_gW = (const float*)d_in[31];
    const float* ro_gb = (const float*)d_in[32];
    float* out = (float*)d_out;

    char* ws = (char*)d_ws;
    size_t off = 0;
    auto wsalloc = [&](size_t bytes) -> void* {
        void* p = ws + off;
        off = (off + bytes + 255) & ~(size_t)255;
        return p;
    };
    _Float16* e8      = (_Float16*)wsalloc((size_t)N_EDGES*8*sizeof(_Float16));
    _Float16* x16     = (_Float16*)wsalloc((size_t)N_NODES*16*sizeof(_Float16));
    float*    inc_sum = (float*)   wsalloc((size_t)N_NODES*64*sizeof(float));
    float*    deg     = (float*)   wsalloc((size_t)N_NODES*sizeof(float));
    float*    e_sum   = (float*)   wsalloc((size_t)NB*64*sizeof(float));
    float*    n_sum   = (float*)   wsalloc((size_t)NB*128*sizeof(float));
    _Float16* Wle     = (_Float16*)wsalloc((size_t)4*32*16*sizeof(_Float16));
    _Float16* Wln     = (_Float16*)wsalloc((size_t)24*32*16*sizeof(_Float16));
    float*    bias_e  = (float*)   wsalloc(64*sizeof(float));
    float*    bias_n  = (float*)   wsalloc(128*sizeof(float));

    hipMemsetAsync(inc_sum, 0, (size_t)N_NODES*64*sizeof(float), stream);
    hipMemsetAsync(deg,     0, (size_t)N_NODES*sizeof(float),    stream);
    hipMemsetAsync(e_sum,   0, (size_t)NB*64*sizeof(float),      stream);
    hipMemsetAsync(n_sum,   0, (size_t)NB*128*sizeof(float),     stream);

    prep_kernel<<<1, 256, 0, stream>>>(g_b, le_We, le_Ws, le_Wg, le_b,
                                       ln_Wn, ln_Wi, ln_Wg, ln_b,
                                       Wle, Wln, bias_e, bias_n);
    edge_enc_kernel<<<(N_EDGES+255)/256, 256, 0, stream>>>(distances, senders, receivers,
                                                           e_W1, e_b1, e_W2, e_b2, e8);
    node_enc_kernel<<<(N_NODES+255)/256, 256, 0, stream>>>(node_features, residues, emb,
                                                           n_W1, n_b1, n_W2, n_b2, x16);
    // 2E/16 = 125000 wave-tiles, 8 waves/block -> 15625 blocks exactly
    edge_mlp_kernel<<<15625, 256, 0, stream>>>(e8, x16, senders, receivers, Wle, bias_e,
                                               inc_sum, e_sum, deg);
    // 100000/16 = 6250 tiles, 8 tiles/block -> 782 blocks (last partial)
    node_mlp_kernel<<<782, 256, 0, stream>>>(x16, inc_sum, deg, Wln, bias_n,
                                             ro_nW, ro_nb, n_sum, out);
    finalize_kernel<<<NB, 32, 0, stream>>>(e_sum, n_sum, g_b, lg_We, lg_Wn, lg_Wg, lg_b,
                                           ro_gW, ro_gb, out + (size_t)N_NODES*2);
}
